// NOLMWA_67345087201538
// MI455X (gfx1250) — matrix operand order repaired and checked by reading, not compiled, not run on hardware
//
#include <hip/hip_runtime.h>
#include <hip/hip_bf16.h>

// ---------------------------------------------------------------------------
// Types for CDNA5 WMMA (wave32): A/B = 16 halves (8 VGPRs), C/D = 8 floats.
// ---------------------------------------------------------------------------
typedef _Float16 h16;
typedef __attribute__((ext_vector_type(16))) _Float16 v16h;
typedef __attribute__((ext_vector_type(8)))  _Float16 v8h;
typedef __attribute__((ext_vector_type(8)))  float    v8f;
typedef __attribute__((ext_vector_type(4)))  unsigned int u32x4;
typedef __attribute__((ext_vector_type(8)))  int      i32x8;
typedef __attribute__((ext_vector_type(4)))  int      i32x4;

// Problem constants (reference is fixed-shape).
#define NN   9856
#define DD   256
#define HD   32
#define HID  512
#define WH   7
#define WW   11
#define LW   77        // tokens per window
#define NWIN 1024      // 8 * 8 * 16
#define MROWS 78848    // B*N = NWIN*LW
#define WGRID 176

// Tensor Data Mover availability (arity differs between toolchains).
#ifdef __has_builtin
#if __has_builtin(__builtin_amdgcn_tensor_load_to_lds) && \
    __has_builtin(__builtin_amdgcn_s_wait_tensorcnt)
#define USE_TDM 1
#endif
#endif
#ifndef USE_TDM
#define USE_TDM 0
#endif

__device__ __forceinline__ v8f wmma_f16(v16h a, v16h b, v8f c) {
  return __builtin_amdgcn_wmma_f32_16x16x32_f16(
      /*neg_a=*/false, a, /*neg_b=*/false, b,
      /*c_mod=*/(short)0, c, /*reuse_a=*/false, /*reuse_b=*/false);
}

union HV { v16h v; v8h h[2]; };

// A-fragment (16x32 f16), ISA layout: lane m=lane&15, ksel=lane>>4;
// halves 0..7  = row[k0 + ksel*8 + 0..7]
// halves 8..15 = row[k0 + 16 + ksel*8 + 0..7]
__device__ __forceinline__ v16h load_afrag(const h16* base, int ld) {
  int lane = threadIdx.x & 31;
  int m = lane & 15, ks = lane >> 4;
  const h16* p = base + m * ld + ks * 8;
  HV u;
  u.h[0] = *(const v8h*)(p);
  u.h[1] = *(const v8h*)(p + 16);
  return u.v;
}

// B-fragment (32x16 f16): column n of B = row n of row-major W (y = x @ W^T).
// lane n=lane&15, ksel=lane>>4; halves 0..7 = Wrow[k0 + ksel*8 + 0..7], 8..15 = Wrow[k0 + 16 + ksel*8 + 0..7]
__device__ __forceinline__ v16h load_bfrag(const h16* base, int ld) {
  int lane = threadIdx.x & 31;
  int n = lane & 15, ks = lane >> 4;
  const h16* p = base + n * ld + ks * 8;
  HV u;
  u.h[0] = *(const v8h*)(p);
  u.h[1] = *(const v8h*)(p + 16);
  return u.v;
}

// ---------------------------------------------------------------------------
// K0: fp32 -> f16 weight convert
// ---------------------------------------------------------------------------
__global__ void f2h_kernel(const float* __restrict__ src, h16* __restrict__ dst, int n) {
  int i = blockIdx.x * 256 + threadIdx.x;
  if (i < n) dst[i] = (h16)src[i];
}

// ---------------------------------------------------------------------------
// K1/K4: LayerNorm over D=256 (one wave per row, 8 elems/lane), optional
// window-partition permute of the destination row. Output f16.
// ---------------------------------------------------------------------------
__global__ __launch_bounds__(256) void ln_kernel(
    const float* __restrict__ x, const float* __restrict__ g,
    const float* __restrict__ bvec, h16* __restrict__ out, int window_permute) {
  int wave = threadIdx.x >> 5;
  int lane = threadIdx.x & 31;
  long row = (long)blockIdx.x * 8 + wave;           // < 78848
  const float* px = x + row * DD + lane * 8;
  float4 f0 = *(const float4*)(px);
  float4 f1 = *(const float4*)(px + 4);
  float v[8] = {f0.x, f0.y, f0.z, f0.w, f1.x, f1.y, f1.z, f1.w};
  float s = 0.f, s2 = 0.f;
#pragma unroll
  for (int i = 0; i < 8; ++i) { s += v[i]; s2 += v[i] * v[i]; }
#pragma unroll
  for (int o = 16; o > 0; o >>= 1) { s += __shfl_xor(s, o); s2 += __shfl_xor(s2, o); }
  float mean = s * (1.0f / DD);
  float var  = s2 * (1.0f / DD) - mean * mean;
  float rstd = rsqrtf(var + 1e-5f);

  long drow = row;
  if (window_permute) {
    int b = (int)(row / NN), n = (int)(row % NN);
    int y = n / WGRID, xc = n % WGRID;
    int ih = y / WH, r = y % WH;
    int iw = xc / WW, c = xc % WW;
    int nw = (b * 8 + ih) * 16 + iw;
    int l = r * WW + c;
    drow = (long)nw * LW + l;
  }
  h16* po = out + drow * DD + lane * 8;
  v8h o8;
#pragma unroll
  for (int i = 0; i < 8; ++i) {
    int d = lane * 8 + i;
    o8[i] = (h16)((v[i] - mean) * rstd * g[d] + bvec[d]);
  }
  *(v8h*)po = o8;
}

// ---------------------------------------------------------------------------
// Phase-1 helper for the attention kernel: one of Q/K/V head projections,
// MAT known at compile time so stores are straight-line (no exec juggling).
// ---------------------------------------------------------------------------
template <int MAT>
__device__ __forceinline__ void qkv_proj(
    const h16 (*xs)[DD], const h16* __restrict__ w_in,
    const float* __restrict__ b_in, int head, int wave, int cn, int chi,
    h16 (*sq)[32], h16 (*sk)[32], h16 (*svT)[96]) {
  for (int job = wave; job < 10; job += 4) {
    int nt = job / 5, mt = job % 5; // nt: 2 col tiles of 16, mt: 5 row tiles
    int row0 = mt * 16;
    int nrow = MAT * DD + head * HD + nt * 16;  // in_proj_w row base
    v8f acc = {0.f, 0.f, 0.f, 0.f, 0.f, 0.f, 0.f, 0.f};
#pragma unroll
    for (int k0 = 0; k0 < DD; k0 += 32) {
      v16h a = load_afrag(&xs[row0][k0], DD);
      v16h b = load_bfrag(w_in + (size_t)nrow * DD + k0, DD);
      acc = wmma_f16(a, b, acc);
    }
    float bias = b_in[nrow + cn];
    int cc = nt * 16 + cn;
    if (MAT == 0) {         // Q, pre-scaled by 1/sqrt(hd)
#pragma unroll
      for (int e = 0; e < 8; ++e)
        sq[row0 + e + 8 * chi][cc] =
            (h16)((acc[e] + bias) * 0.17677669529663687f);
    } else if (MAT == 1) {  // K
#pragma unroll
      for (int e = 0; e < 8; ++e)
        sk[row0 + e + 8 * chi][cc] = (h16)(acc[e] + bias);
    } else {                // V: consecutive e -> consecutive svT columns
      v8h pack;
#pragma unroll
      for (int e = 0; e < 8; ++e) pack[e] = (h16)(acc[e] + bias);
      *(v8h*)&svT[cc][row0 + 8 * chi] = pack;
    }
  }
}

// ---------------------------------------------------------------------------
// K2: fused QKV projection + windowed attention.
// grid = NWIN*HEADS blocks, 128 threads (4 waves). One (window, head) each.
// Window tile staged once into LDS (TDM if available), then all WMMA A
// fragments come from LDS, unguarded (pad rows zeroed).
// LDS overlay: stage buffer (80x256 f16 = 40960B) aliases sp(15360)+sc(25600).
// ---------------------------------------------------------------------------
__global__ __launch_bounds__(128) void attn_kernel(
    const h16* __restrict__ xw, const h16* __restrict__ w_in,
    const float* __restrict__ b_in, h16* __restrict__ attn_out) {
  __shared__ __align__(16) h16 sq[80][32];     // Q (scaled by 1/sqrt(hd))
  __shared__ __align__(16) h16 sk[80][32];     // K
  __shared__ __align__(16) h16 svT[32][96];    // V transposed, K padded to 96
  __shared__ __align__(16) unsigned char ovl[40960];

  h16 (*xs)[DD]  = (h16 (*)[DD])ovl;                 // phases 0-1
  h16 (*sp)[96]  = (h16 (*)[96])ovl;                 // phases 3-4
  float (*sc)[80] = (float (*)[80])(ovl + 15360);    // phases 2-3

  int win  = blockIdx.x >> 3;
  int head = blockIdx.x & 7;
  int wave = threadIdx.x >> 5;
  int lane = threadIdx.x & 31;
  int cn = lane & 15, chi = lane >> 4;

  const h16* xwin = xw + (size_t)win * LW * DD;

  // zero the V padding columns 80..95
  for (int i = threadIdx.x; i < 32 * 16; i += 128)
    svT[i >> 4][80 + (i & 15)] = (h16)0;

  // ---- Phase 0: stage window (77x256 f16) into LDS, pad rows 77..79 ----
#if USE_TDM
  if (wave == 0) {
    unsigned lds_base = (unsigned)(unsigned long long)(&xs[0][0]);
    unsigned long long ga = (unsigned long long)(const void*)xwin;
    u32x4 g0;
    g0[0] = 1u;                                        // count=1 (user mode)
    g0[1] = lds_base;                                  // lds_addr
    g0[2] = (unsigned)(ga & 0xFFFFFFFFu);              // global_addr[31:0]
    g0[3] = (unsigned)((ga >> 32) & 0x1FFFFFFu) | (2u << 30);  // addr hi | type=2
    i32x8 g1;
    g1[0] = (int)(1u << 16);                           // wg_mask=0, data_size=1 (2B)
    g1[1] = (int)((DD & 0xFFFFu) << 16);               // tensor_dim0[15:0] @ [63:48]
    g1[2] = (int)((DD >> 16) | ((unsigned)LW << 16));  // dim0 hi | tensor_dim1 lo
    g1[3] = (int)((LW >> 16) | ((unsigned)DD << 16));  // dim1 hi | tile_dim0
    g1[4] = (int)LW;                                   // tile_dim1 | tile_dim2=0
    g1[5] = (int)DD;                                   // tensor_dim0_stride lo32
    g1[6] = 0;
    g1[7] = 0;
    i32x4 gz = {0, 0, 0, 0};
#if __clang_major__ >= 23
    i32x8 gz8 = {0, 0, 0, 0, 0, 0, 0, 0};
    __builtin_amdgcn_tensor_load_to_lds(g0, g1, gz, gz, gz8, 0);
#else
    __builtin_amdgcn_tensor_load_to_lds(g0, g1, gz, gz, 0);
#endif
    __builtin_amdgcn_s_wait_tensorcnt(0);
  }
  // zero pad rows 77..79 (3 * 256 halves = 96 v8h chunks)
  for (int i = threadIdx.x; i < 96; i += 128) {
    v8h zz = {(h16)0, (h16)0, (h16)0, (h16)0, (h16)0, (h16)0, (h16)0, (h16)0};
    *(v8h*)&xs[LW + (i >> 5)][(i & 31) * 8] = zz;
  }
#else
  for (int idx = threadIdx.x; idx < 80 * 32; idx += 128) {
    int row = idx >> 5, c8 = (idx & 31) * 8;
    v8h val = {(h16)0, (h16)0, (h16)0, (h16)0, (h16)0, (h16)0, (h16)0, (h16)0};
    if (row < LW) val = *(const v8h*)(xwin + (size_t)row * DD + c8);
    *(v8h*)&xs[row][c8] = val;
  }
#endif
  __syncthreads();

  // ---- Phase 1: Q/K/V head projections (3 x 10 WMMA output tiles) ----
  qkv_proj<0>(xs, w_in, b_in, head, wave, cn, chi, sq, sk, svT);
  qkv_proj<1>(xs, w_in, b_in, head, wave, cn, chi, sq, sk, svT);
  qkv_proj<2>(xs, w_in, b_in, head, wave, cn, chi, sq, sk, svT);
  __syncthreads();

  // ---- Phase 2: scores = Q @ K^T (K=32 -> one WMMA per 16x16 tile) ----
  for (int job = wave; job < 25; job += 4) {
    int mt = job / 5, nt = job % 5;
    v8f acc = {0.f, 0.f, 0.f, 0.f, 0.f, 0.f, 0.f, 0.f};
    v16h a = load_afrag(&sq[mt * 16][0], 32);
    v16h b = load_bfrag(&sk[nt * 16][0], 32);
    acc = wmma_f16(a, b, acc);
#pragma unroll
    for (int e = 0; e < 8; ++e)
      sc[mt * 16 + e + 8 * chi][nt * 16 + cn] = acc[e];
  }
  __syncthreads();

  // ---- Phase 3: masked softmax over the 77 real columns ----
  if (threadIdx.x < 80) {
    int r = threadIdx.x;
    float mx = -1e30f;
    for (int c = 0; c < LW; ++c) mx = fmaxf(mx, sc[r][c]);
    float sum = 0.f;
    for (int c = 0; c < LW; ++c) {
      float e = __expf(sc[r][c] - mx);
      sc[r][c] = e;
      sum += e;
    }
    float inv = 1.0f / sum;
    for (int c = 0; c < 96; ++c)
      sp[r][c] = (c < LW) ? (h16)(sc[r][c] * inv) : (h16)0;
  }
  __syncthreads();

  // ---- Phase 4: O = softmax(S) @ V  (K=96 in 3 WMMA steps) ----
  for (int job = wave; job < 10; job += 4) {
    int mt = job / 2, nt = job % 2;
    v8f acc = {0.f, 0.f, 0.f, 0.f, 0.f, 0.f, 0.f, 0.f};
#pragma unroll
    for (int ks = 0; ks < 3; ++ks) {
      v16h a = load_afrag(&sp[mt * 16][ks * 32], 96);
      v16h b = load_bfrag(&svT[nt * 16][ks * 32], 96);
      acc = wmma_f16(a, b, acc);
    }
    int cc = head * HD + nt * 16 + cn;
#pragma unroll
    for (int e = 0; e < 8; ++e) {
      int rr = mt * 16 + e + 8 * chi;
      if (rr < LW)
        attn_out[((size_t)win * LW + rr) * DD + cc] = (h16)acc[e];
    }
  }
}

// ---------------------------------------------------------------------------
// K3/K5/K6: WMMA GEMM. Block tile 128x128 (8 waves), wave tile 64x32
// (4x2 WMMA tiles -> 8 WMMA / 6 fragment loads per k-step, 8 independent
// accumulators for XDL ILP). K templated; k-loop unrolled by 2 for pipelining.
// mode 0: out-proj  -> window-reverse scatter + residual(x) -> d_out (f32)
// mode 1: fc1       -> +bias, exact GELU -> hmid (f16)
// mode 2: fc2       -> +bias + residual(d_out in place) -> d_out (f32)
// ---------------------------------------------------------------------------
template <int K>
__global__ __launch_bounds__(256) void gemm_kernel(
    const h16* __restrict__ A, const h16* __restrict__ W,
    const float* __restrict__ bias,
    const float* resid, float* outf, h16* outh,
    int Nn, int mode) {
  int wave = threadIdx.x >> 5, lane = threadIdx.x & 31;
  int wm = wave & 1, wn = wave >> 1;          // 2 (M) x 4 (N) waves
  long row0 = (long)blockIdx.y * 128 + wm * 64;
  int  col0 = blockIdx.x * 128 + wn * 32;

  v8f z = {0.f, 0.f, 0.f, 0.f, 0.f, 0.f, 0.f, 0.f};
  v8f acc[4][2];
#pragma unroll
  for (int i = 0; i < 4; ++i)
#pragma unroll
    for (int j = 0; j < 2; ++j) acc[i][j] = z;

  const h16* Abase = A + row0 * K;
  const h16* Wbase = W + (size_t)col0 * K;
#pragma unroll 2
  for (int k0 = 0; k0 < K; k0 += 32) {
    __builtin_prefetch(Abase + k0 + 128, 0, 3);
    __builtin_prefetch(Wbase + k0 + 128, 0, 3);
    v16h a[4], b[2];
#pragma unroll
    for (int i = 0; i < 4; ++i)
      a[i] = load_afrag(Abase + (size_t)(i * 16) * K + k0, K);
#pragma unroll
    for (int j = 0; j < 2; ++j)
      b[j] = load_bfrag(Wbase + (size_t)(j * 16) * K + k0, K);
#pragma unroll
    for (int i = 0; i < 4; ++i)
#pragma unroll
      for (int j = 0; j < 2; ++j)
        acc[i][j] = wmma_f16(a[i], b[j], acc[i][j]);
  }

  int cn = lane & 15, chi = lane >> 4;
#pragma unroll
  for (int ti = 0; ti < 4; ++ti) {
#pragma unroll
    for (int tj = 0; tj < 2; ++tj) {
      int ccol = col0 + tj * 16 + cn;
      float bv = bias[ccol];
#pragma unroll
      for (int e = 0; e < 8; ++e) {
        long rr = row0 + ti * 16 + e + 8 * chi;
        float val = acc[ti][tj][e] + bv;
        if (mode == 0) {
          // rr is a window-major row: scatter back to flat (b, n) + residual
          int nw = (int)(rr / LW), l = (int)(rr % LW);
          int b = nw >> 7;
          int ih = (nw >> 4) & 7, iw = nw & 15;
          int r2 = l / WW, c2 = l % WW;
          long n = (long)(ih * WH + r2) * WGRID + iw * WW + c2;
          long idx = ((long)b * NN + n) * DD + ccol;
          outf[idx] = val + resid[idx];
        } else if (mode == 1) {
          float ge = 0.5f * val * (1.0f + erff(val * 0.7071067811865476f));
          outh[rr * (long)Nn + ccol] = (h16)ge;
        } else {
          long idx = rr * (long)DD + ccol;
          outf[idx] = val + resid[idx];   // resid aliases outf: read-then-write
        }
      }
    }
  }
}

// ---------------------------------------------------------------------------
// Host launcher
// ---------------------------------------------------------------------------
extern "C" void kernel_launch(void* const* d_in, const int* in_sizes, int n_in,
                              void* d_out, int out_size, void* d_ws, size_t ws_size,
                              hipStream_t stream) {
  const float* x     = (const float*)d_in[0];
  const float* n1g   = (const float*)d_in[1];
  const float* n1b   = (const float*)d_in[2];
  const float* w_in  = (const float*)d_in[3];
  const float* b_in  = (const float*)d_in[4];
  const float* w_out = (const float*)d_in[5];
  const float* b_out = (const float*)d_in[6];
  const float* n2g   = (const float*)d_in[7];
  const float* n2b   = (const float*)d_in[8];
  const float* w_fc1 = (const float*)d_in[9];
  const float* b_fc1 = (const float*)d_in[10];
  const float* w_fc2 = (const float*)d_in[11];
  const float* b_fc2 = (const float*)d_in[12];
  float* out = (float*)d_out;

  char* ws = (char*)d_ws;
  size_t off = 0;
  auto carve = [&](size_t bytes) {
    size_t cur = off;
    off += (bytes + 255) & ~(size_t)255;
    return cur;
  };
  h16* w_in16  = (h16*)(ws + carve((size_t)3 * DD * DD * 2));
  h16* w_out16 = (h16*)(ws + carve((size_t)DD * DD * 2));
  h16* w_f1_16 = (h16*)(ws + carve((size_t)HID * DD * 2));
  h16* w_f2_16 = (h16*)(ws + carve((size_t)DD * HID * 2));
  h16* xw_h    = (h16*)(ws + carve((size_t)MROWS * DD * 2));
  h16* attn_h  = (h16*)(ws + carve((size_t)MROWS * DD * 2));
  h16* xn2_h   = (h16*)(ws + carve((size_t)MROWS * DD * 2));
  h16* hmid_h  = (h16*)(ws + carve((size_t)MROWS * HID * 2));

  // K0: weight converts
  {
    int n0 = 3 * DD * DD, n1 = DD * DD, n2 = HID * DD, n3 = DD * HID;
    f2h_kernel<<<(n0 + 255) / 256, 256, 0, stream>>>(w_in,  w_in16,  n0);
    f2h_kernel<<<(n1 + 255) / 256, 256, 0, stream>>>(w_out, w_out16, n1);
    f2h_kernel<<<(n2 + 255) / 256, 256, 0, stream>>>(w_fc1, w_f1_16, n2);
    f2h_kernel<<<(n3 + 255) / 256, 256, 0, stream>>>(w_fc2, w_f2_16, n3);
  }

  // K1: LN1 + window partition -> xw_h (window-major, f16)
  ln_kernel<<<MROWS / 8, 256, 0, stream>>>(x, n1g, n1b, xw_h, 1);

  // K2: fused QKV + attention per (window, head) -> attn_h (window-major, f16)
  attn_kernel<<<NWIN * 8, 128, 0, stream>>>(xw_h, w_in16, b_in, attn_h);

  // K3: out-proj + window reverse + residual -> d_out (= x2, f32)
  gemm_kernel<DD><<<dim3(DD / 128, MROWS / 128), 256, 0, stream>>>(
      attn_h, w_out16, b_out, x, out, nullptr, DD, 0);

  // K4: LN2 -> xn2_h (flat, f16)
  ln_kernel<<<MROWS / 8, 256, 0, stream>>>(out, n2g, n2b, xn2_h, 0);

  // K5: FC1 + GELU -> hmid_h (f16)
  gemm_kernel<DD><<<dim3(HID / 128, MROWS / 128), 256, 0, stream>>>(
      xn2_h, w_f1_16, b_fc1, nullptr, nullptr, hmid_h, HID, 1);

  // K6: FC2 + residual (in-place on d_out) -> final output
  gemm_kernel<HID><<<dim3(DD / 128, MROWS / 128), 256, 0, stream>>>(
      hmid_h, w_f2_16, b_fc2, out, out, nullptr, DD, 2);
}
